// FixedPointLSTMCell_52450140619192
// MI455X (gfx1250) — compile-verified, matrix operand order checked by reading, not run on hardware
//
#include <hip/hip_runtime.h>
#include <hip/hip_bf16.h>

typedef __attribute__((ext_vector_type(16))) _Float16 v16h;
typedef __attribute__((ext_vector_type(8)))  _Float16 v8h;
typedef __attribute__((ext_vector_type(4)))  _Float16 v4h;
typedef __attribute__((ext_vector_type(8)))  float    v8f;
typedef __attribute__((ext_vector_type(4)))  float    v4f;

#define BDIM 16384
#define IDIM 512
#define HDIM 512
#define KSTEPS (IDIM / 32)

// Padded LDS tile: 8 quadrants x 16 rows x (32 data + 8 pad) halves.
// Row stride 40 halves = 20 dwords -> lane n fragment reads start at distinct
// banks (20n mod 64 distinct multiples of 4 for n=0..15): conflict-free.
#define LROW 40
#define LBUF (8 * 16 * LROW)   // 5120 halves = 10240 bytes per buffer

// fake_quant: round to 2^-8 grid, clamp to 16-bit fixed-point range
__device__ __forceinline__ float fq(float x) {
    float q = rintf(x * 256.0f);                       // RNE like jnp.round
    q = fminf(fmaxf(q, -32767.0f), 32767.0f);
    return q * 0.00390625f;                            // * 2^-8
}
__device__ __forceinline__ float hsig(float x) {
    return fminf(fmaxf(x * 0.16666667f + 0.5f, 0.0f), 1.0f);
}
__device__ __forceinline__ float htanh(float x) {
    return fminf(fmaxf(x, -1.0f), 1.0f);
}

// Quantize f32 -> fake-quant grid -> f16 (exact for |x| < 8)
__global__ void quant_to_f16_kernel(const float* __restrict__ in,
                                    _Float16* __restrict__ out, int n) {
    int i = (blockIdx.x * blockDim.x + threadIdx.x) * 4;
    if (i >= n) return;
    v4f v = *(const v4f*)(in + i);
    v4h o;
    o[0] = (_Float16)fq(v[0]);
    o[1] = (_Float16)fq(v[1]);
    o[2] = (_Float16)fq(v[2]);
    o[3] = (_Float16)fq(v[3]);
    *(v4h*)(out + i) = o;
}

// 16-bit WMMA operand fragment: per lane, 8 halves at K..K+7 and 8 at K+16..K+23.
__device__ __forceinline__ v16h ldfrag(const _Float16* p) {
    v8h lo = *(const v8h*)(p);
    v8h hi = *(const v8h*)(p + 16);
    return __builtin_shufflevector(lo, hi,
        0, 1, 2, 3, 4, 5, 6, 7, 8, 9, 10, 11, 12, 13, 14, 15);
}

// gfx1250 async global->LDS copy (128b), tracked by ASYNCcnt.
__device__ __forceinline__ void async_load_b128(unsigned lds_off, const void* gptr) {
    asm volatile("global_load_async_to_lds_b128 %0, %1, off"
                 :: "v"(lds_off), "v"(gptr) : "memory");
}

// Workgroup barrier that flushes only DScnt (LDS ordering for buffer reuse).
// Unlike __syncthreads it does NOT drain LOADcnt, so global A-prefetches for
// the next K-step stay in flight across the barrier.
__device__ __forceinline__ void wg_barrier_lds() {
    asm volatile("s_wait_dscnt 0x0\n\t"
                 "s_barrier_signal -1\n\t"
                 "s_barrier_wait -1" ::: "memory");
}

// Fully fused LSTM cell. Per block: 128(M) x 16(N) tile, 8 waves, each wave a
// 16x16 tile with 8 accumulators (all 4 gate quadrants of both GEMMs).
// B tiles triple-buffered in LDS via async loads (1 barrier per K-step);
// A streamed from global, software-pipelined one step ahead.
__global__ void __launch_bounds__(256)
lstm_wmma_kernel(const _Float16* __restrict__ xq,
                 const _Float16* __restrict__ hq,
                 const _Float16* __restrict__ wiq,   // [2048,512] fq'd f16
                 const _Float16* __restrict__ whq,   // [2048,512] fq'd f16
                 const float* __restrict__ b_ih,
                 const float* __restrict__ b_hh,
                 const float* __restrict__ c_prev,
                 float* __restrict__ out_h,
                 float* __restrict__ out_c)
{
    __shared__ __align__(16) _Float16 sB[3][LBUF];

    const int t    = threadIdx.x;
    const int wave = t >> 5;
    const int lane = t & 31;
    const int l15  = lane & 15;
    const int kb   = (lane >> 4) << 3;     // 0 or 8

    const int rowBase = blockIdx.x * 128 + wave * 16;   // M tile (batch)
    const int n0      = blockIdx.y * 16;                // N tile (hidden col)

    const _Float16* xrow = xq + (size_t)(rowBase + l15) * IDIM;
    const _Float16* hrow = hq + (size_t)(rowBase + l15) * HDIM;

    // Per-thread async fill descriptors (2 x 16B chunks of the 8KB B tile per
    // thread), hoisted out of the K-loop: one 64-bit global base each (W_ih /
    // W_hh select resolved once) and one LDS dest register per buffer.
    const _Float16* gsrc[2];
    unsigned ldsd[2][3];
    #pragma unroll
    for (int i = 0; i < 2; ++i) {
        const int c = t + i * 256;          // chunk id 0..511
        const int q = c >> 6;               // quadrant 0..7 (0-3: W_ih, 4-7: W_hh)
        const int r = (c >> 2) & 15;        // row within tile
        const int j = c & 3;                // 16B chunk within 64B row
        gsrc[i] = ((q < 4) ? wiq + (size_t)(q * 512 + n0 + r) * IDIM
                           : whq + (size_t)((q - 4) * 512 + n0 + r) * HDIM) + j * 8;
        const unsigned off = (unsigned)(((q * 16 + r) * LROW + j * 8) * 2);
        ldsd[i][0] = (unsigned)(size_t)&sB[0][0] + off;
        ldsd[i][1] = (unsigned)(size_t)&sB[1][0] + off;
        ldsd[i][2] = (unsigned)(size_t)&sB[2][0] + off;
    }

    v8f acc[8];
    #pragma unroll
    for (int q = 0; q < 8; ++q) { v8f z = {}; acc[q] = z; }

    // Prologue: buffers 0,1 <- K-slices 0,32; prefetch A fragments for step 0.
    async_load_b128(ldsd[0][0], gsrc[0]);
    async_load_b128(ldsd[1][0], gsrc[1]);
    async_load_b128(ldsd[0][1], gsrc[0] + 32);
    async_load_b128(ldsd[1][1], gsrc[1] + 32);
    v16h ax = ldfrag(xrow + kb);
    v16h ah = ldfrag(hrow + kb);

    #pragma unroll
    for (int s = 0; s < KSTEPS; ++s) {
        // Newest outstanding batch is s+1 (2 instructions) -> <=2 means batch s
        // has landed for this wave; for the final step everything must be done.
        if (s < KSTEPS - 1) {
            asm volatile("s_wait_asynccnt 0x2" ::: "memory");
        } else {
            asm volatile("s_wait_asynccnt 0x0" ::: "memory");
        }
        wg_barrier_lds();   // all waves' chunks for buf[s%3] visible; prior ds
                            // reads of buf[(s+2)%3] (step s-1) fenced before refill

        // Refill buffer (s+2)%3 early so the async DMA overlaps this step's WMMAs.
        if (s + 2 < KSTEPS) {
            async_load_b128(ldsd[0][(s + 2) % 3], gsrc[0] + (s + 2) * 32);
            async_load_b128(ldsd[1][(s + 2) % 3], gsrc[1] + (s + 2) * 32);
        }

        // A prefetch for next step; survives the next barrier in flight since
        // wg_barrier_lds does not drain LOADcnt.
        v16h axn, ahn;
        if (s + 1 < KSTEPS) {
            axn = ldfrag(xrow + (s + 1) * 32 + kb);
            ahn = ldfrag(hrow + (s + 1) * 32 + kb);
        }

        const _Float16* bb = &sB[s % 3][0];
        #pragma unroll
        for (int q = 0; q < 4; ++q) {       // 4 x-quadrants, A operand reused
            v16h bi = ldfrag(bb + ((q * 16 + l15) * LROW + kb));
            acc[q] = __builtin_amdgcn_wmma_f32_16x16x32_f16(
                false, ax, false, bi, (short)0, acc[q], false, false);
        }
        #pragma unroll
        for (int q = 0; q < 4; ++q) {       // 4 h-quadrants
            v16h bh = ldfrag(bb + (((q + 4) * 16 + l15) * LROW + kb));
            acc[4 + q] = __builtin_amdgcn_wmma_f32_16x16x32_f16(
                false, ah, false, bh, (short)0, acc[4 + q], false, false);
        }
        if (s + 1 < KSTEPS) { ax = axn; ah = ahn; }
    }

    // Epilogue: C/D layout -> VGPR r, lanes 0-15: M=r; lanes 16-31: M=r+8; N=l15
    const int col  = n0 + l15;
    const int rtop = rowBase + ((lane >> 4) << 3);

    float biq[4], bhq[4];
    #pragma unroll
    for (int q = 0; q < 4; ++q) {
        biq[q] = fq(b_ih[q * HDIM + col]);
        bhq[q] = fq(b_hh[q * HDIM + col]);
    }

    #pragma unroll
    for (int r = 0; r < 8; ++r) {
        const size_t idx = (size_t)(rtop + r) * HDIM + col;
        float gii = fq(acc[0][r] + biq[0]);
        float gif = fq(acc[1][r] + biq[1]);
        float gig = fq(acc[2][r] + biq[2]);
        float gio = fq(acc[3][r] + biq[3]);
        float ghi = fq(acc[4][r] + bhq[0]);
        float ghf = fq(acc[5][r] + bhq[1]);
        float ghg = fq(acc[6][r] + bhq[2]);
        float gho = fq(acc[7][r] + bhq[3]);

        float iv = fq(hsig(gii + ghi));
        float fv = fq(hsig(gif + ghf));
        float gv = fq(htanh(gig + ghg));
        float ov = fq(hsig(gio + gho));

        float cq   = fq(c_prev[idx]);
        float cnew = fv * cq + iv * gv;
        out_c[idx] = cnew;
        out_h[idx] = ov * fq(htanh(cnew));
    }
}

extern "C" void kernel_launch(void* const* d_in, const int* in_sizes, int n_in,
                              void* d_out, int out_size, void* d_ws, size_t ws_size,
                              hipStream_t stream) {
    const float* x      = (const float*)d_in[0];
    const float* h_prev = (const float*)d_in[1];
    const float* c_prev = (const float*)d_in[2];
    const float* W_ih   = (const float*)d_in[3];
    const float* b_ih   = (const float*)d_in[4];
    const float* W_hh   = (const float*)d_in[5];
    const float* b_hh   = (const float*)d_in[6];

    // Workspace: quantized f16 copies of x, h_prev, W_ih, W_hh (~38 MB)
    _Float16* xq  = (_Float16*)d_ws;
    _Float16* hq  = xq  + (size_t)BDIM * IDIM;
    _Float16* wiq = hq  + (size_t)BDIM * HDIM;
    _Float16* whq = wiq + (size_t)4 * HDIM * IDIM;

    const int nx = BDIM * IDIM;       // 8,388,608
    const int nw = 4 * HDIM * IDIM;   // 1,048,576
    quant_to_f16_kernel<<<nx / 4 / 256, 256, 0, stream>>>(x,      xq,  nx);
    quant_to_f16_kernel<<<nx / 4 / 256, 256, 0, stream>>>(h_prev, hq,  nx);
    quant_to_f16_kernel<<<nw / 4 / 256, 256, 0, stream>>>(W_ih,   wiq, nw);
    quant_to_f16_kernel<<<nw / 4 / 256, 256, 0, stream>>>(W_hh,   whq, nw);

    float* out_h = (float*)d_out;
    float* out_c = out_h + (size_t)BDIM * HDIM;

    dim3 grid(BDIM / 128, HDIM / 16);   // 128 x 32 blocks, 256 threads (8 waves)
    lstm_wmma_kernel<<<grid, 256, 0, stream>>>(xq, hq, wiq, whq,
                                               b_ih, b_hh, c_prev, out_h, out_c);
}
